// QuanvolutionEnhanced_65481071401814
// MI455X (gfx1250) — matrix operand order repaired and checked by reading, not compiled, not run on hardware
//
#include <hip/hip_runtime.h>
#include <hip/hip_bf16.h>
#include <stdint.h>

typedef __bf16 bf16_t;
typedef __bf16   v16bf __attribute__((ext_vector_type(16)));
typedef float    v8f   __attribute__((ext_vector_type(8)));
typedef uint32_t u32x4 __attribute__((ext_vector_type(4)));
typedef int      i32x4 __attribute__((ext_vector_type(4)));
typedef int      i32x8 __attribute__((ext_vector_type(8)));

union Frag16 { v16bf v; u32x4 q[2]; };

#if defined(__gfx1250__) && __has_builtin(__builtin_amdgcn_tensor_load_to_lds)
#define HAVE_TDM 1
#endif

// ---------------------------------------------------------------------------
// Weight conversion: fus_w (128,324) f32 -> Bt bf16 [128][352], zero K-pad
// ---------------------------------------------------------------------------
__global__ void wconv_fus(const float* __restrict__ fus_w, bf16_t* __restrict__ Bt) {
    int i = blockIdx.x * 256 + threadIdx.x;
    if (i >= 128 * 352) return;
    int n = i / 352, k = i % 352;
    float val = (k < 324) ? fus_w[n * 324 + k] : 0.0f;
    Bt[i] = (bf16_t)val;
}

// cls_w (10,6272) + reg_w (1,6272) -> Wbf bf16 [16][6272], rows 11..15 zero
__global__ void wconv_fc(const float* __restrict__ cls_w, const float* __restrict__ reg_w,
                         bf16_t* __restrict__ Wbf) {
    int i = blockIdx.x * 256 + threadIdx.x;
    if (i >= 16 * 6272) return;
    int n = i / 6272, k = i % 6272;
    float val = 0.0f;
    if (n < 10)       val = cls_w[n * 6272 + k];
    else if (n == 10) val = reg_w[k];
    Wbf[i] = (bf16_t)val;
}

// ---------------------------------------------------------------------------
// Per-image preprocessing: dw conv -> pw conv(+bias) folded with avgpool,
// quantum patch measurements; emit zero-padded bf16 [36][16][16] tile.
// ---------------------------------------------------------------------------
__global__ void __launch_bounds__(256)
prep_kernel(const float* __restrict__ x,
            const float* __restrict__ dw_w,
            const float* __restrict__ pw_w,
            const float* __restrict__ pw_b,
            const float* __restrict__ U,
            bf16_t* __restrict__ padded36)
{
    __shared__ float ximg[784];
    __shared__ float tconv[784];
    __shared__ float tavg[196];
    __shared__ float Us[256];

    const int b   = blockIdx.x;
    const int tid = threadIdx.x;
    const float* xb = x + (size_t)b * 784;
    bf16_t* pad = padded36 + (size_t)b * 36 * 256;

    for (int i = tid; i < 784; i += 256) ximg[i] = xb[i];
    for (int i = tid; i < 256; i += 256) Us[i]   = U[i];
    for (int i = tid; i < 36 * 256; i += 256) pad[i] = (bf16_t)0.0f;
    __syncthreads();

    // depthwise 3x3, pad 1
    float w9[9];
#pragma unroll
    for (int i = 0; i < 9; ++i) w9[i] = dw_w[i];
    for (int i = tid; i < 784; i += 256) {
        int y = i / 28, xx = i % 28;
        float s = 0.0f;
#pragma unroll
        for (int dy = 0; dy < 3; ++dy)
#pragma unroll
            for (int dx = 0; dx < 3; ++dx) {
                int yy = y + dy - 1, xc = xx + dx - 1;
                if (yy >= 0 && yy < 28 && xc >= 0 && xc < 28)
                    s += w9[dy * 3 + dx] * ximg[yy * 28 + xc];
            }
        tconv[i] = s;
    }
    __syncthreads();

    // 2x2 average of dw output (pw 1x1 commutes with pooling)
    for (int i = tid; i < 196; i += 256) {
        int y = i / 14, xx = i % 14;
        tavg[i] = 0.25f * (tconv[(2*y)*28 + 2*xx]     + tconv[(2*y)*28 + 2*xx + 1] +
                           tconv[(2*y+1)*28 + 2*xx]   + tconv[(2*y+1)*28 + 2*xx + 1]);
    }
    __syncthreads();

    // channels 0..31 : pw conv + bias on pooled map (interior of padded tile)
    for (int i = tid; i < 32 * 196; i += 256) {
        int c = i / 196, p = i % 196;
        int y = p / 14, xx = p % 14;
        float v = pw_w[c] * tavg[p] + pw_b[c];
        pad[c * 256 + (y + 1) * 16 + (xx + 1)] = (bf16_t)v;
    }

    // channels 32..35 : quantum patches (scrambled flat->(4,14,14) layout)
    for (int i = tid; i < 196; i += 256) {
        int y = i / 14, xx = i % 14;
        float h0 = 0.5f * ximg[(2*y)*28 + 2*xx];
        float h1 = 0.5f * ximg[(2*y)*28 + 2*xx + 1];
        float h2 = 0.5f * ximg[(2*y+1)*28 + 2*xx];
        float h3 = 0.5f * ximg[(2*y+1)*28 + 2*xx + 1];
        float a0[2] = {__cosf(h0), __sinf(h0)};
        float a1[2] = {__cosf(h1), __sinf(h1)};
        float a2[2] = {__cosf(h2), __sinf(h2)};
        float a3[2] = {__cosf(h3), __sinf(h3)};
        float amp[16];
#pragma unroll
        for (int i0 = 0; i0 < 2; ++i0)
#pragma unroll
            for (int i1 = 0; i1 < 2; ++i1)
#pragma unroll
                for (int i2 = 0; i2 < 2; ++i2)
#pragma unroll
                    for (int i3 = 0; i3 < 2; ++i3)
                        amp[i0*8 + i1*4 + i2*2 + i3] = a0[i0]*a1[i1]*a2[i2]*a3[i3];
        float meas[4] = {0.f, 0.f, 0.f, 0.f};
#pragma unroll
        for (int s = 0; s < 16; ++s) {
            float st = 0.0f;
#pragma unroll
            for (int t = 0; t < 16; ++t) st += Us[s*16 + t] * amp[t];
            float pr = st * st;
#pragma unroll
            for (int w = 0; w < 4; ++w)
                meas[w] += (((s >> (3 - w)) & 1) ? -pr : pr);
        }
#pragma unroll
        for (int w = 0; w < 4; ++w) {
            int f = i * 4 + w;                 // torch .view scramble
            int cc = f / 196, pos = f % 196;
            int yy = pos / 14, xp = pos % 14;
            pad[(32 + cc) * 256 + (yy + 1) * 16 + (xp + 1)] = (bf16_t)meas[w];
        }
    }
}

// ---------------------------------------------------------------------------
// Fusion conv as implicit-im2col WMMA GEMM per image.
// M=196 (13 tiles), N=128 (8 waves x 16), K=352 (11 chunks of 32, bf16).
// Image tile is fetched by the Tensor Data Mover (async, TENSORcnt) and
// overlapped with the per-wave B-fragment register preload.
// Dynamic LDS: img 18432B | A-tile 11264B | fmap(f32 128x196) 100352B
// ---------------------------------------------------------------------------
__global__ void __launch_bounds__(256)
fusion_wmma_kernel(const bf16_t* __restrict__ padded36,
                   const bf16_t* __restrict__ Bt,
                   const float* __restrict__ fus_b,
                   bf16_t* __restrict__ featbf)
{
    extern __shared__ unsigned char smem[];
    bf16_t* img   = (bf16_t*)smem;             // [36][16][16]
    bf16_t* atile = (bf16_t*)(smem + 18432);   // [16][352]
    float*  fmap  = (float*)(smem + 29696);    // [128][196]

    const int b    = blockIdx.x;
    const int tid  = threadIdx.x;
    const int wave = tid >> 5;
    const int lane = tid & 31;
    const int lhi  = lane >> 4;                // K-half selector
    const int llo  = lane & 15;

#ifdef HAVE_TDM
    // --- TDM: DMA the 18KB image tile global -> LDS (issued by wave 0) ---
    if (wave == 0) {
        uint64_t ga = (uint64_t)(uintptr_t)(padded36 + (size_t)b * 9216);
        u32x4 g0;
        g0[0] = 1u;                                        // count=1 (valid user D#)
        g0[1] = (uint32_t)(uintptr_t)img;                  // lds_addr (bytes)
        g0[2] = (uint32_t)ga;                              // global_addr[31:0]
        g0[3] = ((uint32_t)(ga >> 32) & 0x01FFFFFFu)       // global_addr[56:32]
              | 0x80000000u;                               // type=2 ("image")
        i32x8 g1;
        g1[0] = 0x00020000;          // data_size=2 (4-byte units), no multicast
        g1[1] = (int)(4608u << 16);  // tensor_dim0[15:0]=4608 in bits[63:48]
        g1[2] = (int)(1u << 16);     // tensor_dim0 hi=0 | tensor_dim1=1
        g1[3] = (int)(4608u << 16);  // tensor_dim1 hi=0 | tile_dim0=4608
        g1[4] = 1;                   // tile_dim1=1, tile_dim2=0
        g1[5] = 4608;                // tensor_dim0_stride lo
        g1[6] = 0;                   // stride hi | tensor_dim1_stride lo
        g1[7] = 0;
        i32x4 z4 = {0, 0, 0, 0};
#if __clang_major__ >= 23
        i32x8 z8 = {0, 0, 0, 0, 0, 0, 0, 0};
        __builtin_amdgcn_tensor_load_to_lds(g0, g1, z4, z4, z8, 0);
#else
        __builtin_amdgcn_tensor_load_to_lds(g0, g1, z4, z4, 0);
#endif
    }
#else
    // fallback: cooperative copy (9216 bf16 = 4608 dwords)
    {
        const uint32_t* src = (const uint32_t*)(padded36 + (size_t)b * 9216);
        uint32_t* dst = (uint32_t*)img;
        for (int i = tid; i < 4608; i += 256) dst[i] = src[i];
    }
#endif

    // preload this wave's 11 B fragments (out-channels wave*16 .. +15)
    // -- overlaps with the TDM transfer above
    const int n = wave * 16 + llo;
    Frag16 bfrag[11];
    {
        const bf16_t* bp = Bt + (size_t)n * 352 + (lhi ? 8 : 0);
#pragma unroll
        for (int kc = 0; kc < 11; ++kc) {
            const u32x4* qp = (const u32x4*)(bp + kc * 32);
            bfrag[kc].q[0] = qp[0];
            bfrag[kc].q[1] = qp[2];
        }
    }
    const float bias = fus_b[n];

#ifdef HAVE_TDM
    if (wave == 0) __builtin_amdgcn_s_wait_tensorcnt((short)0);
#endif

    for (int mt = 0; mt < 13; ++mt) {
        __syncthreads();                       // img ready / protect A-tile reuse
        // cooperative im2col build: A[16][352]
        for (int idx = tid; idx < 16 * 352; idx += 256) {
            int m = idx / 352, k = idx % 352;
            bf16_t val = (bf16_t)0.0f;
            int p = mt * 16 + m;
            if (k < 324 && p < 196) {
                int c = k / 9, r = k % 9;
                int dy = r / 3, dx = r % 3;
                int y = p / 14, xx = p % 14;
                val = img[c * 256 + (y + dy) * 16 + (xx + dx)];
            }
            atile[idx] = val;
        }
        __syncthreads();

        v8f acc = {};
        const bf16_t* ap = atile + llo * 352 + (lhi ? 8 : 0);
#pragma unroll
        for (int kc = 0; kc < 11; ++kc) {
            Frag16 af;
            af.q[0] = *(const u32x4*)(ap + kc * 32);
            af.q[1] = *(const u32x4*)(ap + kc * 32 + 16);
            acc = __builtin_amdgcn_wmma_f32_16x16x32_bf16(
                false, af.v, false, bfrag[kc].v, (short)0, acc, false, false);
        }

        // bias + ReLU -> LDS feature map
#pragma unroll
        for (int vi = 0; vi < 8; ++vi) {
            int m = vi + lhi * 8;
            int p = mt * 16 + m;
            if (p < 196) {
                float r = acc[vi] + bias;
                fmap[n * 196 + p] = r > 0.0f ? r : 0.0f;
            }
        }
    }
    __syncthreads();

    // 2x2 avgpool -> bf16 features [128*49] per image (flat NCHW order)
    bf16_t* fout = featbf + (size_t)b * 6272;
    for (int i = tid; i < 6272; i += 256) {
        int c = i / 49, rem = i % 49;
        int py = rem / 7, px = rem % 7;
        int p00 = (2 * py) * 14 + 2 * px;
        const float* row = fmap + c * 196;
        float s = row[p00] + row[p00 + 1] + row[p00 + 14] + row[p00 + 15];
        fout[i] = (bf16_t)(0.25f * s);
    }
}

// ---------------------------------------------------------------------------
// Final FC: M=1024 (64 tiles), N=16 (10 logits + 1 aux + pad), K=6272=196*32.
// One wave per M-tile; 8 waves per block, 8 blocks.
// ---------------------------------------------------------------------------
__global__ void __launch_bounds__(256)
fc_wmma_kernel(const bf16_t* __restrict__ featbf,
               const bf16_t* __restrict__ Wbf,
               const float* __restrict__ cls_b,
               const float* __restrict__ reg_b,
               float* __restrict__ out)
{
    const int tid  = threadIdx.x;
    const int wave = tid >> 5;
    const int lane = tid & 31;
    const int lhi  = lane >> 4;
    const int llo  = lane & 15;
    const int mt   = blockIdx.x * 8 + wave;    // 0..63

    v8f acc = {};
    const bf16_t* ap = featbf + (size_t)(mt * 16 + llo) * 6272 + (lhi ? 8 : 0);
    const bf16_t* bp = Wbf    + (size_t)llo * 6272 + (lhi ? 8 : 0);
    for (int kc = 0; kc < 196; ++kc) {
        Frag16 af, bfr;
        af.q[0]  = *(const u32x4*)(ap + kc * 32);
        af.q[1]  = *(const u32x4*)(ap + kc * 32 + 16);
        bfr.q[0] = *(const u32x4*)(bp + kc * 32);
        bfr.q[1] = *(const u32x4*)(bp + kc * 32 + 16);
        acc = __builtin_amdgcn_wmma_f32_16x16x32_bf16(
            false, af.v, false, bfr.v, (short)0, acc, false, false);
    }

#pragma unroll
    for (int vi = 0; vi < 8; ++vi) {
        int m = vi + lhi * 8;
        int b = mt * 16 + m;
        if (llo < 10)        out[b * 10 + llo] = acc[vi] + cls_b[llo];
        else if (llo == 10)  out[10240 + b]    = acc[vi] + reg_b[0];
    }
}

// ---------------------------------------------------------------------------
extern "C" void kernel_launch(void* const* d_in, const int* in_sizes, int n_in,
                              void* d_out, int out_size, void* d_ws, size_t ws_size,
                              hipStream_t stream)
{
    const float* x     = (const float*)d_in[0];
    const float* dw_w  = (const float*)d_in[1];
    const float* pw_w  = (const float*)d_in[2];
    const float* pw_b  = (const float*)d_in[3];
    const float* U     = (const float*)d_in[4];
    const float* fus_w = (const float*)d_in[5];
    const float* fus_b = (const float*)d_in[6];
    const float* cls_w = (const float*)d_in[7];
    const float* cls_b = (const float*)d_in[8];
    const float* reg_w = (const float*)d_in[9];
    const float* reg_b = (const float*)d_in[10];

    char* ws = (char*)d_ws;
    bf16_t* padded36 = (bf16_t*)(ws);              // 1024*36*256*2  = 18,874,368 B
    bf16_t* Bt       = (bf16_t*)(ws + 18874368);   // 128*352*2     =     90,112 B
    bf16_t* Wbf      = (bf16_t*)(ws + 18964480);   // 16*6272*2     =    200,704 B
    bf16_t* featbf   = (bf16_t*)(ws + 19165184);   // 1024*6272*2   = 12,845,056 B

    wconv_fus<<<(128 * 352 + 255) / 256, 256, 0, stream>>>(fus_w, Bt);
    wconv_fc <<<(16 * 6272 + 255) / 256, 256, 0, stream>>>(cls_w, reg_w, Wbf);
    prep_kernel<<<1024, 256, 0, stream>>>(x, dw_w, pw_w, pw_b, U, padded36);
    fusion_wmma_kernel<<<1024, 256, 130048, stream>>>(padded36, Bt, fus_b, featbf);
    fc_wmma_kernel<<<8, 256, 0, stream>>>(featbf, Wbf, cls_b, reg_b, (float*)d_out);
}
